// UniTransformerO2TwoUpdateGeneral_66391604461755
// MI455X (gfx1250) — matrix operand order repaired and checked by reading, not compiled, not run on hardware
//
#include <hip/hip_runtime.h>
#include <hip/hip_bf16.h>

// ---------------------------------------------------------------------------
// UniTransformerO2TwoUpdateGeneral for MI455X (gfx1250, wave32, WMMA)
//
// Sizes (fixed by the reference):
//   N=30000 nodes, E=480000 edges, H=128, NH=16, HD=8, SD=32, EF=4, NG=20
//   KV = 2H + EF + NG*EF + SD = 372  (padded to 384 for K%32==0)
//
// All dense layers run as 16-row tiles through v_wmma_f32_16x16x32_f16:
//   one wave per tile, A staged in LDS (f16), B pre-repacked into WMMA
//   fragment order (contiguous v16h per lane), f32 accumulators, LN/ReLU
//   between the two GEMMs done via LDS.
// Scatter softmax: order-preserving uint atomicMax + float atomicAdd.
// ---------------------------------------------------------------------------

#define NN 30000
#define EE 480000
#define HH 128
#define NHH 16
#define KVP 384        // padded KV
#define KVA 372        // actual KV

typedef __attribute__((ext_vector_type(16))) _Float16 v16h;
typedef __attribute__((ext_vector_type(8)))  _Float16 v8h;
typedef __attribute__((ext_vector_type(8)))  float    v8f;

// ---------------- WMMA helpers -------------------------------------------

__device__ inline v8f vzero8() {
  v8f z;
#pragma unroll
  for (int j = 0; j < 8; ++j) z[j] = 0.f;
  return z;
}

// A fragment: 16x32 f16 tile row-major in LDS, stride ldk halves.
// ISA layout: lane<16 holds row=lane, K = kb+0..7 (elems 0..7) and
// kb+16..23 (elems 8..15) with kb = (lane<16 ? 0 : 8).
__device__ inline v16h load_a_frag(const _Float16* sA, int lane, int kt, int ldk) {
  int row = lane & 15;
  int kb  = kt * 32 + ((lane < 16) ? 0 : 8);
  const _Float16* p = sA + row * ldk + kb;
  v8h lo = *(const v8h*)(p);
  v8h hi = *(const v8h*)(p + 16);
  return __builtin_shufflevector(lo, hi, 0,1,2,3,4,5,6,7,8,9,10,11,12,13,14,15);
}

// B fragments pre-packed: index = (((kt*NT)+nt)*32 + lane)*16 halves.
template<int NT>
__device__ inline void wmma_gemm(const _Float16* sA, int ldk, int ksteps,
                                 const _Float16* Wpk, v8f* acc, int lane) {
  for (int kt = 0; kt < ksteps; ++kt) {
    v16h a = load_a_frag(sA, lane, kt, ldk);
    const _Float16* wb = Wpk + (((size_t)kt * NT) * 32 + (size_t)lane) * 16;
    __builtin_prefetch(wb + (size_t)NT * 32 * 16, 0, 3);
#pragma unroll
    for (int nt = 0; nt < NT; ++nt) {
      v16h b = *(const v16h*)(wb + (size_t)nt * 32 * 16);
      acc[nt] = __builtin_amdgcn_wmma_f32_16x16x32_f16(
          /*neg_a=*/false, a, /*neg_b=*/false, b,
          /*c_mod=*/(short)0, acc[nt], /*reuse_a=*/false, /*reuse_b=*/false);
    }
  }
}

// C/D layout: VGPR r, lane l -> row M = (l<16 ? r : 8+r), col N = nt*16 + (l&15)
template<int NT>
__device__ inline void acc_to_smem(const v8f* acc, float* sY, int ldn,
                                   const float* bias, int lane) {
  int col0 = lane & 15;
  int rb   = (lane < 16) ? 0 : 8;
#pragma unroll
  for (int nt = 0; nt < NT; ++nt) {
    int col = nt * 16 + col0;
    float bv = bias[col];
#pragma unroll
    for (int r = 0; r < 8; ++r) sY[(rb + r) * ldn + col] = acc[nt][r] + bv;
  }
}

// LayerNorm(128) + ReLU, f32 LDS rows -> f16 LDS rows (lanes 0..15, 1 row each)
__device__ inline void ln_relu_row(const float* sY, _Float16* sA2,
                                   const float* g, const float* be, int lane) {
  if (lane < 16) {
    const float* y = sY + lane * HH;
    float mu = 0.f;
    for (int i = 0; i < HH; ++i) mu += y[i];
    mu *= (1.f / 128.f);
    float var = 0.f;
    for (int i = 0; i < HH; ++i) { float t = y[i] - mu; var += t * t; }
    var *= (1.f / 128.f);
    float inv = rsqrtf(var + 1e-5f);
    _Float16* o = sA2 + lane * HH;
    for (int i = 0; i < HH; ++i) {
      float t = (y[i] - mu) * inv * g[i] + be[i];
      o[i] = (_Float16)fmaxf(t, 0.f);
    }
  }
}

// ---------------- weight repack (f32 row-major [K,N] -> WMMA frag f16) ----

__global__ void repack_w_kernel(const float* __restrict__ W, _Float16* __restrict__ out,
                                int Kpad, int Kact, int Ncols) {
  int idx = blockIdx.x * blockDim.x + threadIdx.x;
  int NT = Ncols / 16;
  int total = (Kpad / 32) * NT * 32 * 16;
  if (idx >= total) return;
  int j    = idx & 15;
  int lane = (idx >> 4) & 31;
  int fn   = idx >> 9;          // kt*NT + nt
  int nt   = fn % NT;
  int kt   = fn / NT;
  int col  = nt * 16 + (lane & 15);
  int krow = kt * 32 + ((lane < 16) ? 0 : 8) + ((j < 8) ? j : 8 + j);
  float v = (krow < Kact) ? W[(size_t)krow * Ncols + col] : 0.f;
  out[idx] = (_Float16)v;
}

// ---------------- node MLP: [concat src0,src1] -> 128 -> LN -> ReLU -> 128 -

__global__ __launch_bounds__(32) void node_mlp_kernel(
    const float* __restrict__ src0, int w0,
    const float* __restrict__ src1, int w1, int Kpad,
    const _Float16* __restrict__ W1pk, const float* __restrict__ b1,
    const float* __restrict__ g, const float* __restrict__ be,
    const _Float16* __restrict__ W2pk, const float* __restrict__ b2,
    const float* __restrict__ res, float* __restrict__ out) {
  __shared__ __align__(16) _Float16 sA[16 * 256];
  __shared__ __align__(16) float    sY[16 * HH];
  __shared__ __align__(16) _Float16 sA2[16 * HH];
  int lane = threadIdx.x;
  int n0 = blockIdx.x * 16;
  if (lane < 16) {
    int n = n0 + lane;
    _Float16* row = sA + lane * Kpad;
    for (int i = 0; i < w0; ++i) row[i] = (_Float16)src0[(size_t)n * w0 + i];
    for (int i = 0; i < w1; ++i) row[w0 + i] = (_Float16)src1[(size_t)n * w1 + i];
    for (int i = w0 + w1; i < Kpad; ++i) row[i] = (_Float16)0.f;
  }
  __syncthreads();
  v8f acc[8];
#pragma unroll
  for (int i = 0; i < 8; ++i) acc[i] = vzero8();
  wmma_gemm<8>(sA, Kpad, Kpad / 32, W1pk, acc, lane);
  acc_to_smem<8>(acc, sY, HH, b1, lane);
  __syncthreads();
  ln_relu_row(sY, sA2, g, be, lane);
  __syncthreads();
#pragma unroll
  for (int i = 0; i < 8; ++i) acc[i] = vzero8();
  wmma_gemm<8>(sA2, HH, 4, W2pk, acc, lane);
  int col0 = lane & 15, rb = (lane < 16) ? 0 : 8;
#pragma unroll
  for (int nt = 0; nt < 8; ++nt) {
    int col = nt * 16 + col0;
    float bv = b2[col];
#pragma unroll
    for (int r = 0; r < 8; ++r) {
      int n = n0 + rb + r;
      float v = acc[nt][r] + bv;
      if (res) v += res[(size_t)n * HH + col];
      out[(size_t)n * HH + col] = v;
    }
  }
}

// ---------------- KV tile build (shared by both edge layers) --------------

__device__ inline void build_kv_row(_Float16* row, int e,
                                    const float* __restrict__ x,
                                    const float* __restrict__ edge_attr,
                                    const float* __restrict__ hfeat,
                                    const float* __restrict__ inv_shape,
                                    const int* __restrict__ src,
                                    const int* __restrict__ dst) {
  int s = src[e], d = dst[e];
  float rx = x[d * 3 + 0] - x[s * 3 + 0];
  float ry = x[d * 3 + 1] - x[s * 3 + 1];
  float rz = x[d * 3 + 2] - x[s * 3 + 2];
  float dist = sqrtf(rx * rx + ry * ry + rz * rz);
  float ea[4];
#pragma unroll
  for (int a = 0; a < 4; ++a) { ea[a] = edge_attr[(size_t)e * 4 + a]; row[a] = (_Float16)ea[a]; }
  const float coeff = -1.805f;                // -0.5 / (10/19)^2
  const float step  = 10.f / 19.f;
  for (int gi = 0; gi < 20; ++gi) {
    float t = dist - step * (float)gi;
    float gs = __expf(coeff * t * t);
#pragma unroll
    for (int a = 0; a < 4; ++a) row[4 + a * 20 + gi] = (_Float16)(ea[a] * gs);
  }
  for (int i = 0; i < HH; ++i) row[84 + i]  = (_Float16)hfeat[(size_t)d * HH + i];
  for (int i = 0; i < HH; ++i) row[212 + i] = (_Float16)hfeat[(size_t)s * HH + i];
  for (int i = 0; i < 32; ++i) row[340 + i] = (_Float16)inv_shape[(size_t)d * 32 + i];
  for (int i = KVA; i < KVP; ++i) row[i] = (_Float16)0.f;
}

// ---------------- edge X2H: kv -> k (logits) and v ------------------------

__global__ __launch_bounds__(32) void edge_x2h_kernel(
    const float* __restrict__ h, const float* __restrict__ x,
    const float* __restrict__ edge_attr, const float* __restrict__ inv_shape,
    const float* __restrict__ e_w, const int* __restrict__ src,
    const int* __restrict__ dst, const float* __restrict__ q,
    const _Float16* Wk1, const float* bk1, const float* gk, const float* bek,
    const _Float16* Wk2, const float* bk2,
    const _Float16* Wv1, const float* bv1, const float* gv, const float* bev,
    const _Float16* Wv2, const float* bv2,
    float* __restrict__ logits, _Float16* __restrict__ vbuf) {
  __shared__ __align__(16) _Float16 sKV[16 * KVP];
  __shared__ __align__(16) float    sY[16 * HH];
  __shared__ __align__(16) _Float16 sA2[16 * HH];
  int lane = threadIdx.x;
  int e0 = blockIdx.x * 16;
  if (lane < 16)
    build_kv_row(sKV + lane * KVP, e0 + lane, x, edge_attr, h, inv_shape, src, dst);
  __syncthreads();

  v8f acc[8];
  // ---- k = MLP_hk(kv) ----
#pragma unroll
  for (int i = 0; i < 8; ++i) acc[i] = vzero8();
  wmma_gemm<8>(sKV, KVP, KVP / 32, Wk1, acc, lane);
  acc_to_smem<8>(acc, sY, HH, bk1, lane);
  __syncthreads();
  ln_relu_row(sY, sA2, gk, bek, lane);
  __syncthreads();
#pragma unroll
  for (int i = 0; i < 8; ++i) acc[i] = vzero8();
  wmma_gemm<8>(sA2, HH, 4, Wk2, acc, lane);
  acc_to_smem<8>(acc, sY, HH, bk2, lane);
  __syncthreads();
  // logits[e][head] = (q[dst]*k).sum(headdim) / sqrt(8)
  if (lane < 16) {
    int e = e0 + lane, d = dst[e];
    const float sc = 0.35355339059f;
    for (int hh = 0; hh < NHH; ++hh) {
      float a = 0.f;
      for (int j = 0; j < 8; ++j)
        a += sY[lane * HH + hh * 8 + j] * q[(size_t)d * HH + hh * 8 + j];
      logits[(size_t)e * NHH + hh] = a * sc;
    }
  }
  __syncthreads();
  // ---- v = MLP_hv(kv) * e_w ----
#pragma unroll
  for (int i = 0; i < 8; ++i) acc[i] = vzero8();
  wmma_gemm<8>(sKV, KVP, KVP / 32, Wv1, acc, lane);
  acc_to_smem<8>(acc, sY, HH, bv1, lane);
  __syncthreads();
  ln_relu_row(sY, sA2, gv, bev, lane);
  __syncthreads();
#pragma unroll
  for (int i = 0; i < 8; ++i) acc[i] = vzero8();
  wmma_gemm<8>(sA2, HH, 4, Wv2, acc, lane);
  int col0 = lane & 15, rb = (lane < 16) ? 0 : 8;
#pragma unroll
  for (int nt = 0; nt < 8; ++nt) {
    int col = nt * 16 + col0;
    float bv = bv2[col];
#pragma unroll
    for (int r = 0; r < 8; ++r) {
      int e = e0 + rb + r;
      vbuf[(size_t)e * HH + col] = (_Float16)((acc[nt][r] + bv) * e_w[e]);
    }
  }
}

// ---------------- edge H2X: kv2 -> k2 (logits2) and v2raw -----------------

__global__ __launch_bounds__(32) void edge_h2x_kernel(
    const float* __restrict__ hn, const float* __restrict__ x,
    const float* __restrict__ edge_attr, const float* __restrict__ inv_shape,
    const float* __restrict__ e_w, const int* __restrict__ src,
    const int* __restrict__ dst, const float* __restrict__ q2,
    const _Float16* Wk1, const float* bk1, const float* gk, const float* bek,
    const _Float16* Wk2, const float* bk2,
    const _Float16* Wv1, const float* bv1, const float* gv, const float* bev,
    const _Float16* Wv2, const float* bv2,
    float* __restrict__ logits, float* __restrict__ v2raw) {
  __shared__ __align__(16) _Float16 sKV[16 * KVP];
  __shared__ __align__(16) float    sY[16 * HH];
  __shared__ __align__(16) _Float16 sA2[16 * HH];
  int lane = threadIdx.x;
  int e0 = blockIdx.x * 16;
  if (lane < 16)
    build_kv_row(sKV + lane * KVP, e0 + lane, x, edge_attr, hn, inv_shape, src, dst);
  __syncthreads();

  v8f acc[8];
  // ---- k2 = MLP_xk(kv2) ----
#pragma unroll
  for (int i = 0; i < 8; ++i) acc[i] = vzero8();
  wmma_gemm<8>(sKV, KVP, KVP / 32, Wk1, acc, lane);
  acc_to_smem<8>(acc, sY, HH, bk1, lane);
  __syncthreads();
  ln_relu_row(sY, sA2, gk, bek, lane);
  __syncthreads();
#pragma unroll
  for (int i = 0; i < 8; ++i) acc[i] = vzero8();
  wmma_gemm<8>(sA2, HH, 4, Wk2, acc, lane);
  acc_to_smem<8>(acc, sY, HH, bk2, lane);
  __syncthreads();
  if (lane < 16) {
    int e = e0 + lane, d = dst[e];
    const float sc = 0.35355339059f;
    for (int hh = 0; hh < NHH; ++hh) {
      float a = 0.f;
      for (int j = 0; j < 8; ++j)
        a += sY[lane * HH + hh * 8 + j] * q2[(size_t)d * HH + hh * 8 + j];
      logits[(size_t)e * NHH + hh] = a * sc;
    }
  }
  __syncthreads();
  // ---- v2 = MLP_xv(kv2) * e_w  (out dim = 16 heads) ----
#pragma unroll
  for (int i = 0; i < 8; ++i) acc[i] = vzero8();
  wmma_gemm<8>(sKV, KVP, KVP / 32, Wv1, acc, lane);
  acc_to_smem<8>(acc, sY, HH, bv1, lane);
  __syncthreads();
  ln_relu_row(sY, sA2, gv, bev, lane);
  __syncthreads();
  v8f acc1 = vzero8();
  wmma_gemm<1>(sA2, HH, 4, Wv2, &acc1, lane);
  int head = lane & 15, rb = (lane < 16) ? 0 : 8;
  float bv = bv2[head];
#pragma unroll
  for (int r = 0; r < 8; ++r) {
    int e = e0 + rb + r;
    v2raw[(size_t)e * NHH + head] = (acc1[r] + bv) * e_w[e];
  }
}

// ---------------- scatter softmax helpers ---------------------------------

__device__ inline unsigned fenc(float f) {
  unsigned u = __float_as_uint(f);
  return (u & 0x80000000u) ? ~u : (u | 0x80000000u);
}
__device__ inline float fdec(unsigned e) {
  unsigned u = (e & 0x80000000u) ? (e & 0x7fffffffu) : ~e;
  return __uint_as_float(u);
}

__global__ void seg_max_kernel(const float* __restrict__ logits,
                               const int* __restrict__ dst,
                               unsigned* __restrict__ maxenc) {
  long long i = (long long)blockIdx.x * blockDim.x + threadIdx.x;
  if (i >= (long long)EE * NHH) return;
  int e = (int)(i / NHH), hh = (int)(i % NHH);
  atomicMax(&maxenc[(size_t)dst[e] * NHH + hh], fenc(logits[i]));
}

__global__ void seg_expsum_kernel(float* __restrict__ logits,
                                  const int* __restrict__ dst,
                                  const unsigned* __restrict__ maxenc,
                                  float* __restrict__ sums) {
  long long i = (long long)blockIdx.x * blockDim.x + threadIdx.x;
  if (i >= (long long)EE * NHH) return;
  int e = (int)(i / NHH), hh = (int)(i % NHH);
  size_t mi = (size_t)dst[e] * NHH + hh;
  float ex = __expf(logits[i] - fdec(maxenc[mi]));
  logits[i] = ex;
  atomicAdd(&sums[mi], ex);
}

__global__ void agg_x2h_kernel(const float* __restrict__ logits,
                               const float* __restrict__ sums,
                               const _Float16* __restrict__ vbuf,
                               const int* __restrict__ dst,
                               float* __restrict__ agg) {
  long long i = (long long)blockIdx.x * blockDim.x + threadIdx.x;
  if (i >= (long long)EE * HH) return;
  int e = (int)(i / HH), f = (int)(i % HH), hh = f >> 3;
  int d = dst[e];
  float alpha = logits[(size_t)e * NHH + hh] /
                (sums[(size_t)d * NHH + hh] + 1e-16f);
  atomicAdd(&agg[(size_t)d * HH + f], alpha * (float)vbuf[i]);
}

__global__ void agg_h2x_kernel(const float* __restrict__ logits,
                               const float* __restrict__ sums,
                               const float* __restrict__ v2raw,
                               const float* __restrict__ x,
                               const int* __restrict__ src,
                               const int* __restrict__ dst,
                               float* __restrict__ agg2) {
  long long i = (long long)blockIdx.x * blockDim.x + threadIdx.x;
  if (i >= (long long)EE * NHH) return;
  int e = (int)(i / NHH), hh = (int)(i % NHH);
  int s = src[e], d = dst[e];
  float alpha = logits[i] / (sums[(size_t)d * NHH + hh] + 1e-16f);
  float val = alpha * v2raw[i];
#pragma unroll
  for (int c = 0; c < 3; ++c) {
    float rel = x[d * 3 + c] - x[s * 3 + c];
    atomicAdd(&agg2[(size_t)d * 48 + hh * 3 + c], val * rel);
  }
}

// ---------------- final: VNLinearLeakyReLU + coordinate update ------------

__global__ void finalize_kernel(const float* __restrict__ x,
                                const float* __restrict__ agg2,
                                const float* __restrict__ shape_emb,
                                const float* __restrict__ Wf,
                                const float* __restrict__ Wd,
                                float* __restrict__ xnew) {
  int n = blockIdx.x * blockDim.x + threadIdx.x;
  if (n >= NN) return;
  float outsum[3] = {0.f, 0.f, 0.f}, vnsum[3] = {0.f, 0.f, 0.f};
  for (int o = 0; o < NHH; ++o) {
    float pf[3] = {0.f, 0.f, 0.f}, dd[3] = {0.f, 0.f, 0.f};
    for (int c = 0; c < 49; ++c) {
      float wf = Wf[c * NHH + o], wd = Wd[c * NHH + o];
#pragma unroll
      for (int d3 = 0; d3 < 3; ++d3) {
        float t;
        if (c == 0)       t = x[n * 3 + d3];
        else if (c < 17)  t = agg2[(size_t)n * 48 + (c - 1) * 3 + d3];
        else              t = shape_emb[(size_t)n * 96 + (c - 17) * 3 + d3];
        pf[d3] += t * wf;
        dd[d3] += t * wd;
      }
    }
    float dot = pf[0] * dd[0] + pf[1] * dd[1] + pf[2] * dd[2];
    float dsq = dd[0] * dd[0] + dd[1] * dd[1] + dd[2] * dd[2];
    float sc = dot / (dsq + 1e-6f);
#pragma unroll
    for (int d3 = 0; d3 < 3; ++d3) {
      float alt = (dot >= 0.f) ? pf[d3] : (pf[d3] - sc * dd[d3]);
      vnsum[d3] += 0.2f * pf[d3] + 0.8f * alt;
      outsum[d3] += agg2[(size_t)n * 48 + o * 3 + d3];
    }
  }
#pragma unroll
  for (int d3 = 0; d3 < 3; ++d3)
    xnew[n * 3 + d3] = x[n * 3 + d3] + outsum[d3] * (1.f / 16.f) + vnsum[d3] * (1.f / 16.f);
}

// ---------------------------------------------------------------------------
// Host-side launch orchestration.
// Input order: setup_inputs() insertion order for arrays, then `params`
// flattened as a JAX pytree (sorted dict keys):
//   0 h, 1 x, 2 edge_attr, 3 invar_ligand_shape, 4 ligand_shape_emb,
//   5 e_w, 6 edge_index,
//   7 h2x/Wd, 8 h2x/Wf,
//   9..14  h2x/xk {W1,W2,b1,b2,be,g}
//  15..20  h2x/xq {W1,W2,b1,b2,be,g}
//  21..26  h2x/xv {W1,W2,b1,b2,be,g}
//  27..32  x2h/hk {W1,W2,b1,b2,be,g}
//  33..38  x2h/hq {W1,W2,b1,b2,be,g}
//  39..44  x2h/hv {W1,W2,b1,b2,be,g}
//  45..50  x2h/out{W1,W2,b1,b2,be,g}
// d_out: [N*128] x2h_out (=hn) then [N*3] x_new.
// ---------------------------------------------------------------------------

struct MlpP { const float *W1, *W2, *b1, *b2, *be, *g; };

extern "C" void kernel_launch(void* const* d_in, const int* in_sizes, int n_in,
                              void* d_out, int out_size, void* d_ws, size_t ws_size,
                              hipStream_t stream) {
  (void)in_sizes; (void)n_in; (void)out_size; (void)ws_size;

  const float* h   = (const float*)d_in[0];
  const float* x   = (const float*)d_in[1];
  const float* ea  = (const float*)d_in[2];
  const float* inv = (const float*)d_in[3];
  const float* she = (const float*)d_in[4];
  const float* ew  = (const float*)d_in[5];
  const int*   ei  = (const int*)d_in[6];
  const int* srcp = ei;
  const int* dstp = ei + EE;
  const float* Wd = (const float*)d_in[7];
  const float* Wf = (const float*)d_in[8];
  auto mlp = [&](int base) {
    MlpP p;
    p.W1 = (const float*)d_in[base + 0];
    p.W2 = (const float*)d_in[base + 1];
    p.b1 = (const float*)d_in[base + 2];
    p.b2 = (const float*)d_in[base + 3];
    p.be = (const float*)d_in[base + 4];
    p.g  = (const float*)d_in[base + 5];
    return p;
  };
  MlpP xk = mlp(9),  xq = mlp(15), xv = mlp(21);
  MlpP hk = mlp(27), hq = mlp(33), hv = mlp(39), po = mlp(45);

  float* hn   = (float*)d_out;            // x2h_out == hn
  float* xnew = (float*)d_out + (size_t)NN * HH;

  // ---- workspace carve-out ----
  char* w = (char*)d_ws;
  auto alloc = [&](size_t bytes) {
    void* p = (void*)w;
    w += (bytes + 255) & ~(size_t)255;
    return p;
  };
  _Float16* pWhk1 = (_Float16*)alloc(sizeof(_Float16) * KVP * HH);
  _Float16* pWhv1 = (_Float16*)alloc(sizeof(_Float16) * KVP * HH);
  _Float16* pWxk1 = (_Float16*)alloc(sizeof(_Float16) * KVP * HH);
  _Float16* pWxv1 = (_Float16*)alloc(sizeof(_Float16) * KVP * HH);
  _Float16* pWhk2 = (_Float16*)alloc(sizeof(_Float16) * HH * HH);
  _Float16* pWhv2 = (_Float16*)alloc(sizeof(_Float16) * HH * HH);
  _Float16* pWxk2 = (_Float16*)alloc(sizeof(_Float16) * HH * HH);
  _Float16* pWhq1 = (_Float16*)alloc(sizeof(_Float16) * HH * HH);
  _Float16* pWhq2 = (_Float16*)alloc(sizeof(_Float16) * HH * HH);
  _Float16* pWxq1 = (_Float16*)alloc(sizeof(_Float16) * HH * HH);
  _Float16* pWxq2 = (_Float16*)alloc(sizeof(_Float16) * HH * HH);
  _Float16* pWo1  = (_Float16*)alloc(sizeof(_Float16) * 256 * HH);
  _Float16* pWo2  = (_Float16*)alloc(sizeof(_Float16) * HH * HH);
  _Float16* pWxv2 = (_Float16*)alloc(sizeof(_Float16) * HH * 16);
  float*    q     = (float*)alloc(sizeof(float) * (size_t)NN * HH);
  float*    q2    = (float*)alloc(sizeof(float) * (size_t)NN * HH);
  float*    agg   = (float*)alloc(sizeof(float) * (size_t)NN * HH);
  float*    agg2  = (float*)alloc(sizeof(float) * (size_t)NN * 48);
  unsigned* mxe   = (unsigned*)alloc(sizeof(unsigned) * (size_t)NN * NHH);
  float*    sums  = (float*)alloc(sizeof(float) * (size_t)NN * NHH);
  float*    logit = (float*)alloc(sizeof(float) * (size_t)EE * NHH);
  _Float16* vbuf  = (_Float16*)alloc(sizeof(_Float16) * (size_t)EE * HH);
  float*    v2raw = (float*)alloc(sizeof(float) * (size_t)EE * NHH);

  // ---- 1) repack weights to WMMA fragment order (f16) ----
  auto rp = [&](const float* W, _Float16* o, int Kpad, int Kact, int Ncols) {
    int total = (Kpad / 32) * (Ncols / 16) * 32 * 16;
    repack_w_kernel<<<(total + 255) / 256, 256, 0, stream>>>(W, o, Kpad, Kact, Ncols);
  };
  rp(hk.W1, pWhk1, KVP, KVA, HH);  rp(hk.W2, pWhk2, HH, HH, HH);
  rp(hv.W1, pWhv1, KVP, KVA, HH);  rp(hv.W2, pWhv2, HH, HH, HH);
  rp(xk.W1, pWxk1, KVP, KVA, HH);  rp(xk.W2, pWxk2, HH, HH, HH);
  rp(xv.W1, pWxv1, KVP, KVA, HH);  rp(xv.W2, pWxv2, HH, HH, 16);
  rp(hq.W1, pWhq1, HH, HH, HH);    rp(hq.W2, pWhq2, HH, HH, HH);
  rp(xq.W1, pWxq1, HH, HH, HH);    rp(xq.W2, pWxq2, HH, HH, HH);
  rp(po.W1, pWo1, 256, 256, HH);   rp(po.W2, pWo2, HH, HH, HH);

  const int nodeTiles = NN / 16;   // 1875
  const int edgeTiles = EE / 16;   // 30000
  const long long EH  = (long long)EE * NHH;
  const long long EF_ = (long long)EE * HH;

  // ---- 2) q = MLP_hq(h) ----
  node_mlp_kernel<<<nodeTiles, 32, 0, stream>>>(
      h, HH, nullptr, 0, HH, pWhq1, hq.b1, hq.g, hq.be, pWhq2, hq.b2, nullptr, q);

  // ---- 3) edge X2H: k->logits, v ----
  edge_x2h_kernel<<<edgeTiles, 32, 0, stream>>>(
      h, x, ea, inv, ew, srcp, dstp, q,
      pWhk1, hk.b1, hk.g, hk.be, pWhk2, hk.b2,
      pWhv1, hv.b1, hv.g, hv.be, pWhv2, hv.b2, logit, vbuf);

  // ---- 4) scatter softmax + aggregate ----
  hipMemsetAsync(mxe, 0, sizeof(unsigned) * (size_t)NN * NHH, stream);
  hipMemsetAsync(sums, 0, sizeof(float) * (size_t)NN * NHH, stream);
  seg_max_kernel<<<(int)((EH + 255) / 256), 256, 0, stream>>>(logit, dstp, mxe);
  seg_expsum_kernel<<<(int)((EH + 255) / 256), 256, 0, stream>>>(logit, dstp, mxe, sums);
  hipMemsetAsync(agg, 0, sizeof(float) * (size_t)NN * HH, stream);
  agg_x2h_kernel<<<(int)((EF_ + 255) / 256), 256, 0, stream>>>(logit, sums, vbuf, dstp, agg);

  // ---- 5) x2h_out = MLP_out([agg, h]) + h  (written to d_out as hn) ----
  node_mlp_kernel<<<nodeTiles, 32, 0, stream>>>(
      agg, HH, h, HH, 256, pWo1, po.b1, po.g, po.be, pWo2, po.b2, h, hn);

  // ---- 6) q2 = MLP_xq(hn) ----
  node_mlp_kernel<<<nodeTiles, 32, 0, stream>>>(
      hn, HH, nullptr, 0, HH, pWxq1, xq.b1, xq.g, xq.be, pWxq2, xq.b2, nullptr, q2);

  // ---- 7) edge H2X: k2->logits2, v2raw ----
  edge_h2x_kernel<<<edgeTiles, 32, 0, stream>>>(
      hn, x, ea, inv, ew, srcp, dstp, q2,
      pWxk1, xk.b1, xk.g, xk.be, pWxk2, xk.b2,
      pWxv1, xv.b1, xv.g, xv.be, pWxv2, xv.b2, logit, v2raw);

  // ---- 8) scatter softmax2 + equivariant aggregate ----
  hipMemsetAsync(mxe, 0, sizeof(unsigned) * (size_t)NN * NHH, stream);
  hipMemsetAsync(sums, 0, sizeof(float) * (size_t)NN * NHH, stream);
  seg_max_kernel<<<(int)((EH + 255) / 256), 256, 0, stream>>>(logit, dstp, mxe);
  seg_expsum_kernel<<<(int)((EH + 255) / 256), 256, 0, stream>>>(logit, dstp, mxe, sums);
  hipMemsetAsync(agg2, 0, sizeof(float) * (size_t)NN * 48, stream);
  agg_h2x_kernel<<<(int)((EH + 255) / 256), 256, 0, stream>>>(
      logit, sums, v2raw, x, srcp, dstp, agg2);

  // ---- 9) VN-LeakyReLU + x update ----
  finalize_kernel<<<(NN + 255) / 256, 256, 0, stream>>>(x, agg2, she, Wf, Wd, xnew);
}
